// PretrainedMambaFusionCore_64209761075743
// MI455X (gfx1250) — compile-verified, matrix operand order checked by reading, not run on hardware
//
#include <hip/hip_runtime.h>
#include <math.h>

// ---------------- problem constants (match reference) ----------------
#define HD    1024          // H
#define DI    2048          // inner dim
#define SEQ   1024          // S
#define NB    2             // BATCH
#define ROWS  (NB*SEQ)      // 2048 token rows
#define NS    16            // N state
#define KCONV 4
#define EPSLN 1e-5f
#define PSTR  36            // padded proj row stride (33 -> 36 floats = 9 x b128)

// ---------------- WMMA types ----------------
typedef __attribute__((ext_vector_type(16))) __bf16          v16bf;
typedef __attribute__((ext_vector_type(8)))  float           v8f;
typedef __attribute__((ext_vector_type(8)))  unsigned short  v8us;

union Frag {            // build bf16 fragment from two 16-byte LDS chunks
  v8us  half[2];
  v16bf bf;
};

__device__ __forceinline__ unsigned short f2bf(float f) {
  unsigned int u = __float_as_uint(f);
  u += 0x7fffu + ((u >> 16) & 1u);   // round-to-nearest-even
  return (unsigned short)(u >> 16);
}
__device__ __forceinline__ float silu_f(float x) { return x / (1.f + __expf(-x)); }
__device__ __forceinline__ float sigm_f(float x) { return 1.f / (1.f + __expf(-x)); }

// ---------------- f32 -> bf16 bulk convert (weights / activations) -----------
__global__ __launch_bounds__(256)
void cvt_bf16_kernel(const float* __restrict__ in, unsigned short* __restrict__ out,
                     long n) {
  const long i = ((long)blockIdx.x * 256 + threadIdx.x) * 4;
  if (i + 3 < n) {
    const float4 v = *(const float4*)(in + i);
    ushort4 o;
    o.x = f2bf(v.x); o.y = f2bf(v.y); o.z = f2bf(v.z); o.w = f2bf(v.w);
    *(ushort4*)(out + i) = o;
  }
}

// ---------------- double-buffered bf16 WMMA GEMM: C[M,N] = A[M,K] * W[N,K]^T ----
// Tiles staged memory->LDS with GLOBAL_LOAD_ASYNC_TO_LDS_B128 (no VGPR staging,
// no spills); ASYNCcnt + barrier per K-step. block tile 128x128, K-step 32,
// 8 waves as 2(M)x4(N); wave tile 64x32 = 4x2 WMMA.
// EPI: 0 none | 1 +resid | 2 +bias,gelu(exact) | 3 +bias+resid
// OUTB: store bf16 (C16) instead of f32 (C)
#define BM 128
#define BN 128
#define BK 32
#define LDS_S 40   // ushort stride per row (16B-aligned chunks, conflict-free)

template<int EPI, bool OUTB>
__global__ __launch_bounds__(256)
void gemm_wmma_kernel(const unsigned short* __restrict__ A, int lda,
                      const unsigned short* __restrict__ W, int ldw,
                      float* __restrict__ C, unsigned short* __restrict__ C16, int ldc,
                      int K,
                      const float* __restrict__ bias,
                      const float* __restrict__ resid, int ldr) {
  __shared__ unsigned short Lds[2][(BM + BN) * LDS_S];

  const int bm   = blockIdx.y * BM;
  const int bn   = blockIdx.x * BN;
  const int t    = threadIdx.x;
  const int wid  = t >> 5;
  const int lane = t & 31;
  const int wm   = wid >> 2;            // 0..1 -> 64-row strip
  const int wn   = wid & 3;             // 0..3 -> 32-col strip
  const int lr   = lane & 15;
  const int hi   = (lane & 16) >> 1;    // 0 or 8 (K-half select per ISA layout)

  const int r0 = t >> 2;                // 0..63
  const int c0 = (t & 3) << 3;          // 0,8,16,24 (bf16 elems)

  // per-lane LDS byte offsets of the 4 staged b128 chunks (within one buffer)
  const unsigned ldsbase = (unsigned)(unsigned long long)(void*)&Lds[0][0];
  const unsigned bufStride = (unsigned)((BM + BN) * LDS_S * 2);   // bytes
  const unsigned offA0 = (unsigned)(((r0)       * LDS_S + c0) * 2);
  const unsigned offA1 = (unsigned)(((r0 + 64)  * LDS_S + c0) * 2);
  const unsigned offW0 = (unsigned)(((BM + r0)      * LDS_S + c0) * 2);
  const unsigned offW1 = (unsigned)(((BM + r0 + 64) * LDS_S + c0) * 2);

  // per-lane global source pointers (advance by BK bf16 = 64B per K-step)
  const unsigned short* gA0 = A + (long)(bm + r0)      * lda + c0;
  const unsigned short* gA1 = A + (long)(bm + r0 + 64) * lda + c0;
  const unsigned short* gW0 = W + (long)(bn + r0)      * ldw + c0;
  const unsigned short* gW1 = W + (long)(bn + r0 + 64) * ldw + c0;

  v8f acc[4][2] = {};

  auto asyncTile = [&](int buf) {
    const unsigned base = ldsbase + (unsigned)buf * bufStride;
    asm volatile("global_load_async_to_lds_b128 %0, %1, off"
                 :: "v"(base + offA0), "v"(gA0) : "memory");
    asm volatile("global_load_async_to_lds_b128 %0, %1, off"
                 :: "v"(base + offA1), "v"(gA1) : "memory");
    asm volatile("global_load_async_to_lds_b128 %0, %1, off"
                 :: "v"(base + offW0), "v"(gW0) : "memory");
    asm volatile("global_load_async_to_lds_b128 %0, %1, off"
                 :: "v"(base + offW1), "v"(gW1) : "memory");
  };
  auto compute = [&](int buf) {
    Frag a[4], b[2];
    #pragma unroll
    for (int mi = 0; mi < 4; ++mi) {
      const unsigned short* p = &Lds[buf][(wm * 64 + mi * 16 + lr) * LDS_S];
      a[mi].half[0] = *(const v8us*)(p + hi);
      a[mi].half[1] = *(const v8us*)(p + 16 + hi);
    }
    #pragma unroll
    for (int ni = 0; ni < 2; ++ni) {
      const unsigned short* p = &Lds[buf][(BM + wn * 32 + ni * 16 + lr) * LDS_S];
      b[ni].half[0] = *(const v8us*)(p + hi);
      b[ni].half[1] = *(const v8us*)(p + 16 + hi);
    }
    #pragma unroll
    for (int mi = 0; mi < 4; ++mi)
      #pragma unroll
      for (int ni = 0; ni < 2; ++ni)
        acc[mi][ni] = __builtin_amdgcn_wmma_f32_16x16x32_bf16(
            false, a[mi].bf, false, b[ni].bf, (short)0, acc[mi][ni], false, false);
  };

  // prologue: stage tile 0
  asyncTile(0);
  asm volatile("s_wait_asynccnt 0" ::: "memory");
  __syncthreads();

  int cur = 0;
  for (int k0 = BK; k0 < K; k0 += BK) {
    gA0 += BK; gA1 += BK; gW0 += BK; gW1 += BK;
    asyncTile(cur ^ 1);   // async copies for next tile (overlap with WMMA)
    compute(cur);         // WMMA from current buffer
    asm volatile("s_wait_asynccnt 0" ::: "memory");
    __syncthreads();
    cur ^= 1;
  }
  compute(cur);

  // epilogue + store (C layout: vgpr i -> row i + hi, col = lane&15)
  #pragma unroll
  for (int mi = 0; mi < 4; ++mi)
    #pragma unroll
    for (int ni = 0; ni < 2; ++ni) {
      const int col = bn + wn * 32 + ni * 16 + lr;
      #pragma unroll
      for (int i = 0; i < 8; ++i) {
        const int row = bm + wm * 64 + mi * 16 + hi + i;
        float v = acc[mi][ni][i];
        if (EPI == 1) v += resid[(long)row * ldr + col];
        if (EPI == 2) { v += bias[col]; v = 0.5f * v * (1.f + erff(v * 0.70710678118654752f)); }
        if (EPI == 3) v += bias[col] + resid[(long)row * ldr + col];
        if (OUTB) C16[(long)row * ldc + col] = f2bf(v);
        else      C  [(long)row * ldc + col] = v;
      }
    }
}

// ---------------- layernorm (optional gate epilogue; f32 or bf16 out) --------
template<bool GATE, bool OUTB>
__global__ __launch_bounds__(256)
void ln_kernel(const float* __restrict__ in, const float* __restrict__ g,
               const float* __restrict__ b, const float* __restrict__ gate,
               int gate_stride, int gate_off,
               float* __restrict__ outf, unsigned short* __restrict__ outb, int Cdim) {
  __shared__ float s1[256], s2[256];
  const int row = blockIdx.x;
  const float* x = in + (long)row * Cdim;
  float ls = 0.f, lq = 0.f;
  for (int i = threadIdx.x; i < Cdim; i += 256) { float v = x[i]; ls += v; lq += v * v; }
  s1[threadIdx.x] = ls; s2[threadIdx.x] = lq;
  __syncthreads();
  for (int off = 128; off > 0; off >>= 1) {
    if (threadIdx.x < off) { s1[threadIdx.x] += s1[threadIdx.x + off]; s2[threadIdx.x] += s2[threadIdx.x + off]; }
    __syncthreads();
  }
  const float mu  = s1[0] / (float)Cdim;
  const float var = s2[0] / (float)Cdim - mu * mu;
  const float rstd = rsqrtf(var + EPSLN);
  for (int i = threadIdx.x; i < Cdim; i += 256) {
    float v = (x[i] - mu) * rstd * g[i] + b[i];
    if (GATE) v *= sigm_f(gate[(long)row * gate_stride + gate_off + i]);
    if (OUTB) outb[(long)row * Cdim + i] = f2bf(v);
    else      outf[(long)row * Cdim + i] = v;
  }
}

// ---------------- causal depthwise conv K=4, fused silu ----------------
template<bool INSILU>
__global__ __launch_bounds__(256)
void conv_kernel(const float* __restrict__ in, int in_stride, int in_off,
                 const float* __restrict__ w, const float* __restrict__ bias,
                 float* __restrict__ out) {
  const long idx = (long)blockIdx.x * 256 + threadIdx.x;   // (b*SEQ+s)*DI + d
  const int d = (int)(idx & (DI - 1));
  const int s = (int)((idx >> 11) & (SEQ - 1));
  const int b = (int)(idx >> 21);
  float acc = bias[d];
  #pragma unroll
  for (int k = 0; k < KCONV; ++k) {
    const int sp = s - (KCONV - 1) + k;
    if (sp >= 0) {
      float v = in[((long)(b * SEQ + sp)) * in_stride + in_off + d];
      if (INSILU) v = silu_f(v);
      acc += w[d * KCONV + k] * v;
    }
  }
  out[idx] = silu_f(acc);
}

// ---------------- x_proj: [ROWS,DI] x [33,DI]^T -> [ROWS, PSTR-padded 33] ------
__global__ __launch_bounds__(64)
void xproj_kernel(const float* __restrict__ xin, const float* __restrict__ w,
                  float* __restrict__ proj) {
  const int row = blockIdx.x;
  const int p = threadIdx.x;
  if (p >= 1 + 2 * NS) return;
  const float* xr = xin + (long)row * DI;
  const float* wr = w + (long)p * DI;
  float acc = 0.f;
  for (int k = 0; k < DI; k += 4)
    acc += xr[k] * wr[k] + xr[k + 1] * wr[k + 1] + xr[k + 2] * wr[k + 2] + xr[k + 3] * wr[k + 3];
  proj[(long)row * PSTR + p] = acc;
}

// ---------------- selective scan: 4096 channels, state in registers -----------
// Per-step dt/B/C broadcast staged memory->LDS with async copies (ASYNCcnt).
__global__ __launch_bounds__(256)
void scan_kernel(const float* __restrict__ proj, const float* __restrict__ xin,
                 const float* __restrict__ dtw, const float* __restrict__ dtb,
                 const float* __restrict__ Dv, float* __restrict__ y) {
  __shared__ float sp[PSTR];
  const int d = (blockIdx.x & 7) * 256 + threadIdx.x;  // 0..2047
  const int b = blockIdx.x >> 3;                       // 0..1
  float h[NS];
  #pragma unroll
  for (int n = 0; n < NS; ++n) h[n] = 0.f;
  const float w = dtw[d], bb = dtb[d], Dd = Dv[d];
  const unsigned spbase = (unsigned)(unsigned long long)(void*)&sp[0]; // LDS offset = addr[31:0]
  for (int s = 0; s < SEQ; ++s) {
    __syncthreads();  // previous step's LDS reads are consumed before overwrite
    if (threadIdx.x < PSTR / 4) {   // 9 lanes, all in wave 0
      const float* g = proj + ((long)(b * SEQ + s)) * PSTR + threadIdx.x * 4;
      const unsigned lds = spbase + threadIdx.x * 16;
      asm volatile("global_load_async_to_lds_b128 %0, %1, off"
                   :: "v"(lds), "v"(g) : "memory");
    }
    asm volatile("s_wait_asynccnt 0" ::: "memory");  // no-op for non-issuing waves
    __syncthreads();
    const float dtr = sp[0] * w + bb;
    const float dt = (dtr > 20.f) ? dtr : log1pf(__expf(dtr));   // softplus
    const float xt = xin[((long)(b * SEQ + s)) * DI + d];
    float acc = 0.f;
    #pragma unroll
    for (int n = 0; n < NS; ++n) {
      const float An = -(float)(n + 1);
      h[n] += dt * (h[n] * An + xt * sp[1 + n]);
      acc += h[n] * sp[1 + NS + n];
    }
    y[((long)(b * SEQ + s)) * DI + d] = acc + xt * Dd;           // + x_inner*D fused
  }
}

// ---------------- orchestration ----------------
extern "C" void kernel_launch(void* const* d_in, const int* in_sizes, int n_in,
                              void* d_out, int out_size, void* d_ws, size_t ws_size,
                              hipStream_t stream) {
  const float* hs        = (const float*)d_in[0];
  const float* in_g      = (const float*)d_in[1];
  const float* in_b      = (const float*)d_in[2];
  const float* lng       = (const float*)d_in[3];
  const float* lnb       = (const float*)d_in[4];
  const float* in_proj_w = (const float*)d_in[5];
  const float* conv_w    = (const float*)d_in[6];
  const float* conv_b    = (const float*)d_in[7];
  const float* x_proj_w  = (const float*)d_in[8];
  const float* dt_proj_w = (const float*)d_in[9];
  const float* dt_proj_b = (const float*)d_in[10];
  const float* Dp        = (const float*)d_in[11];
  const float* mln_g     = (const float*)d_in[12];
  const float* mln_b     = (const float*)d_in[13];
  const float* out_proj_w= (const float*)d_in[14];
  const float* op1_w     = (const float*)d_in[15];
  const float* op1_b     = (const float*)d_in[16];
  const float* op2_w     = (const float*)d_in[17];
  const float* op2_b     = (const float*)d_in[18];

  // ---------------- workspace layout ----------------
  // f32 region
  float* ws = (float*)d_ws;
  float* X  = ws;                                // [ROWS, H]
  float* XR = X  + (size_t)ROWS * HD;            // [ROWS, 2*DI]
  float* B0 = XR + (size_t)ROWS * 2 * DI;        // [ROWS, DI]
  float* B1 = B0 + (size_t)ROWS * DI;            // [ROWS, DI]
  float* Y  = B1 + (size_t)ROWS * DI;            // [ROWS, DI]
  float* PR = Y  + (size_t)ROWS * DI;            // [ROWS, PSTR]
  // bf16 region
  unsigned short* u = (unsigned short*)(PR + (size_t)ROWS * PSTR);
  unsigned short* IPW16 = u;                               // 4*4096*1024
  unsigned short* OPW16 = IPW16 + (size_t)4 * 2 * DI * HD; // 4*1024*2048
  unsigned short* O1W16 = OPW16 + (size_t)4 * HD * DI;     // 1024*1024
  unsigned short* O2W16 = O1W16 + (size_t)HD * HD;         // 1024*1024
  unsigned short* XN16  = O2W16 + (size_t)HD * HD;         // [ROWS, H]
  unsigned short* B116  = XN16  + (size_t)ROWS * HD;       // [ROWS, DI]
  unsigned short* X16   = B116  + (size_t)ROWS * DI;       // [ROWS, H]
  unsigned short* T16   = X16   + (size_t)ROWS * HD;       // [ROWS, H]

  // ---- weight preconversion to bf16 (once per launch; amortized vs L2 reads)
  {
    const long n_ip = (long)4 * 2 * DI * HD;
    const long n_op = (long)4 * HD * DI;
    const long n_o  = (long)HD * HD;
    cvt_bf16_kernel<<<(unsigned)(n_ip / 1024), 256, 0, stream>>>(in_proj_w,  IPW16, n_ip);
    cvt_bf16_kernel<<<(unsigned)(n_op / 1024), 256, 0, stream>>>(out_proj_w, OPW16, n_op);
    cvt_bf16_kernel<<<(unsigned)(n_o  / 1024), 256, 0, stream>>>(op1_w, O1W16, n_o);
    cvt_bf16_kernel<<<(unsigned)(n_o  / 1024), 256, 0, stream>>>(op2_w, O2W16, n_o);
  }

  // x = LN(hidden)  (f32: feeds next LN + residual path)
  ln_kernel<false, false><<<ROWS, 256, 0, stream>>>(hs, in_g, in_b, nullptr, 0, 0,
                                                    X, nullptr, HD);

  for (int l = 0; l < 4; ++l) {
    // xn = LN(x)  -> bf16 (only consumed as GEMM A-operand)
    ln_kernel<false, true><<<ROWS, 256, 0, stream>>>(X, lng + l * HD, lnb + l * HD,
                                                     nullptr, 0, 0, nullptr, XN16, HD);
    // xr = xn @ in_proj_w^T   [2048 x 4096]  (f32 out: conv + gating consume it)
    gemm_wmma_kernel<0, false><<<dim3(2 * DI / BN, ROWS / BM), 256, 0, stream>>>(
        XN16, HD, IPW16 + (size_t)l * 2 * DI * HD, HD, XR, nullptr, 2 * DI,
        HD, nullptr, nullptr, 0);
    // x_inner = silu(conv(silu(xr[:, :DI])));  x_inner = silu(conv(x_inner))
    conv_kernel<true ><<<(ROWS * DI) / 256, 256, 0, stream>>>(
        XR, 2 * DI, 0, conv_w + (size_t)(l * 2 + 0) * DI * KCONV,
        conv_b + (size_t)(l * 2 + 0) * DI, B1);
    conv_kernel<false><<<(ROWS * DI) / 256, 256, 0, stream>>>(
        B1, DI, 0, conv_w + (size_t)(l * 2 + 1) * DI * KCONV,
        conv_b + (size_t)(l * 2 + 1) * DI, B0);
    // proj = x_inner @ x_proj_w^T  [2048 x 33] (stride-36 rows)
    xproj_kernel<<<ROWS, 64, 0, stream>>>(B0, x_proj_w + (size_t)l * (1 + 2 * NS) * DI, PR);
    // selective scan (+ x_inner*D fused)
    scan_kernel<<<16, 256, 0, stream>>>(PR, B0, dt_proj_w + l * DI, dt_proj_b + l * DI,
                                        Dp + l * DI, Y);
    // y = LN(y) * sigmoid(res)  -> bf16 (only consumed as GEMM A-operand)
    ln_kernel<true, true><<<ROWS, 256, 0, stream>>>(Y, mln_g + l * DI, mln_b + l * DI,
                                                    XR, 2 * DI, DI, nullptr, B116, DI);
    // x = x + y_g @ out_proj_w^T (residual fused, in-place on X)
    gemm_wmma_kernel<1, false><<<dim3(HD / BN, ROWS / BM), 256, 0, stream>>>(
        B116, DI, OPW16 + (size_t)l * HD * DI, DI, X, nullptr, HD,
        DI, nullptr, X, HD);
  }

  // bf16 copy of final x for the MLP A-operand
  cvt_bf16_kernel<<<(ROWS * HD) / 1024, 256, 0, stream>>>(X, X16, (long)ROWS * HD);
  // t = gelu(x @ op1_w^T + b1) -> bf16 (only consumed by op2)
  gemm_wmma_kernel<2, true><<<dim3(HD / BN, ROWS / BM), 256, 0, stream>>>(
      X16, HD, O1W16, HD, nullptr, T16, HD, HD, op1_b, nullptr, 0);
  // out = t @ op2_w^T + b2 + hidden  (f32 output)
  gemm_wmma_kernel<3, false><<<dim3(HD / BN, ROWS / BM), 256, 0, stream>>>(
      T16, HD, O2W16, HD, (float*)d_out, nullptr, HD, HD, op2_b, hs, HD);
}
